// ChannelAttention_55954833932510
// MI455X (gfx1250) — matrix operand order repaired and checked by reading, not compiled, not run on hardware
//
#include <hip/hip_runtime.h>
#include <hip/hip_bf16.h>

// Problem constants
#define BB 16
#define CC 64
#define NN 512
#define TT 128
#define LDS_COLS ((size_t)NN * TT)   // 65536: row stride of signals[b, c, :, :] flattened

typedef __attribute__((ext_vector_type(16))) __bf16 v16bf;
typedef __attribute__((ext_vector_type(8)))  float  v8f;

// ---------------------------------------------------------------------------
// Kernel 1: k[b,c,t] = sum_i alpha[i] * signals[b,c,i,t]
// grid = B*C blocks, 128 threads (t). Coalesced: consecutive t across lanes.
// ---------------------------------------------------------------------------
__global__ __launch_bounds__(TT) void ca_reduce_k(const float* __restrict__ sig,
                                                  const float* __restrict__ alpha,
                                                  float* __restrict__ kout) {
    const int bc = blockIdx.x;            // 0..B*C-1
    const int t  = threadIdx.x;           // 0..127
    const float* p = sig + (size_t)bc * NN * TT + t;
    float acc = 0.0f;
#pragma unroll 8
    for (int i = 0; i < NN; ++i) {
        acc = fmaf(alpha[i], p[(size_t)i * TT], acc);   // alpha[i] uniform -> s_load
    }
    kout[(size_t)bc * TT + t] = acc;
}

// ---------------------------------------------------------------------------
// Kernel 2: scores[b,c,d] = k[b,c,:] @ Wc @ k[b,d,:], softmax over d -> att
// grid = B blocks, 256 threads.
// ---------------------------------------------------------------------------
__global__ __launch_bounds__(256) void ca_scores_softmax(const float* __restrict__ kin,
                                                         const float* __restrict__ Wc,
                                                         float* __restrict__ att) {
    __shared__ float s_k[CC * TT];    // 32 KB
    __shared__ float s_kt[CC * TT];   // 32 KB  (64 KB total; gfx1250 WGP has 320 KB LDS)

    const int b   = blockIdx.x;
    const int tid = threadIdx.x;
    const float* kb = kin + (size_t)b * CC * TT;

    // Stage k[b] in LDS (coalesced)
    for (int e = tid; e < CC * TT; e += 256) s_k[e] = kb[e];
    __syncthreads();

    // kt[c,s] = sum_t k[c,t] * Wc[t,s]   (8192 outputs, 32 per thread)
    for (int e = 0; e < 32; ++e) {
        const int idx = tid + e * 256;
        const int c = idx >> 7;
        const int s = idx & 127;
        float acc = 0.0f;
#pragma unroll 4
        for (int t = 0; t < TT; ++t)
            acc = fmaf(s_k[c * TT + t], Wc[t * TT + s], acc);
        s_kt[idx] = acc;
    }
    __syncthreads();

    // scores row r = tid>>2; this thread owns 16 columns: d = (tid&3)*16 .. +15
    const int r = tid >> 2;
    const int g = tid & 3;
    float sc[16];
#pragma unroll
    for (int dd = 0; dd < 16; ++dd) {
        const int d = g * 16 + dd;
        float acc = 0.0f;
#pragma unroll 4
        for (int t = 0; t < TT; ++t)
            acc = fmaf(s_kt[r * TT + t], s_k[d * TT + t], acc);
        sc[dd] = acc;
    }

    // Softmax across the 64 columns of row r: 4 consecutive lanes share a row.
    float m = sc[0];
#pragma unroll
    for (int dd = 1; dd < 16; ++dd) m = fmaxf(m, sc[dd]);
    m = fmaxf(m, __shfl_xor(m, 1, 4));
    m = fmaxf(m, __shfl_xor(m, 2, 4));

    float sum = 0.0f;
#pragma unroll
    for (int dd = 0; dd < 16; ++dd) {
        sc[dd] = __expf(sc[dd] - m);
        sum += sc[dd];
    }
    sum += __shfl_xor(sum, 1, 4);
    sum += __shfl_xor(sum, 2, 4);
    const float inv = 1.0f / sum;

    float* arow = att + (size_t)b * CC * CC + (size_t)r * CC + g * 16;
#pragma unroll
    for (int dd = 0; dd < 16; ++dd) arow[dd] = sc[dd] * inv;
}

// ---------------------------------------------------------------------------
// Kernel 3: out[b] (64 x 65536) = att[b] (64x64) x signals[b] (64 x 65536)
// 8 waves/block; each wave computes a 64-row x 16-col strip per iteration
// using v_wmma_f32_16x16x32_bf16 (K=64 -> 2 WMMA per 16x16 tile).
// ---------------------------------------------------------------------------
__global__ __launch_bounds__(256) void ca_mix_wmma(const float* __restrict__ sig,
                                                   const float* __restrict__ att,
                                                   float* __restrict__ out) {
    const int b    = blockIdx.y;
    const int lane = threadIdx.x & 31;
    const int wave = threadIdx.x >> 5;   // 0..7
    const int half = lane >> 4;          // 0 | 1
    const int lcol = lane & 15;          // column-in-tile / row-in-tile index

    __shared__ float s_att[CC * CC];     // 16 KB
    const float* attb = att + (size_t)b * CC * CC;
    for (int e = threadIdx.x; e < CC * CC; e += 256) s_att[e] = attb[e];
    __syncthreads();

    // Build the 8 A-operands (4 M-tiles x 2 K-chunks of 32) once per block.
    // 16-bit A 16x32 layout (ISA 7.12.2): both lane halves hold M=0..15;
    // VGPR v=0..3 -> K = half*8 + {2v,2v+1}; v=4..7 -> K = 16 + half*8 + {..}.
    v16bf A[4][2];
#pragma unroll
    for (int mt = 0; mt < 4; ++mt) {
        const int m = mt * 16 + lcol;
#pragma unroll
        for (int kc = 0; kc < 2; ++kc) {
            const int kbase = kc * 32;
            v16bf a;
#pragma unroll
            for (int v = 0; v < 4; ++v) {
                const int k0 = kbase + half * 8 + 2 * v;
                a[2 * v]     = (__bf16)s_att[m * CC + k0];
                a[2 * v + 1] = (__bf16)s_att[m * CC + k0 + 1];
            }
#pragma unroll
            for (int v = 0; v < 4; ++v) {
                const int k0 = kbase + 16 + half * 8 + 2 * v;
                a[8 + 2 * v]     = (__bf16)s_att[m * CC + k0];
                a[8 + 2 * v + 1] = (__bf16)s_att[m * CC + k0 + 1];
            }
            A[mt][kc] = a;
        }
    }

    const float* Sb = sig + (size_t)b * CC * LDS_COLS;
    float*       Ob = out + (size_t)b * CC * LDS_COLS;

    // 512 strips of 128 columns per batch; each wave takes 16 of the 128.
    for (int si = blockIdx.x; si < 512; si += gridDim.x) {
        const int j0 = si * 128 + wave * 16;

        // B operand, 32x16 bf16 (same K order as A: VGPR v=0..3 -> K = half*8 + {2v,2v+1};
        // v=4..7 -> K = 16 + half*8 + {2(v-4), 2(v-4)+1}) for column N = lane&15.
        v16bf Bop[2];
#pragma unroll
        for (int kc = 0; kc < 2; ++kc) {
            v16bf bb;
#pragma unroll
            for (int v = 0; v < 8; ++v) {
                const int k = kc * 32 + (v >> 2) * 16 + half * 8 + 2 * (v & 3);
                bb[2 * v]     = (__bf16)Sb[(size_t)k * LDS_COLS + j0 + lcol];
                bb[2 * v + 1] = (__bf16)Sb[(size_t)(k + 1) * LDS_COLS + j0 + lcol];
            }
            Bop[kc] = bb;
        }

#pragma unroll
        for (int mt = 0; mt < 4; ++mt) {
            v8f c = {};
            c = __builtin_amdgcn_wmma_f32_16x16x32_bf16(false, A[mt][0], false, Bop[0],
                                                        (short)0, c, false, false);
            c = __builtin_amdgcn_wmma_f32_16x16x32_bf16(false, A[mt][1], false, Bop[1],
                                                        (short)0, c, false, false);
            // D layout: VGPR r -> (M = mt*16 + r + half*8, N = j0 + lcol)
#pragma unroll
            for (int r = 0; r < 8; ++r) {
                const int row = mt * 16 + r + half * 8;
                Ob[(size_t)row * LDS_COLS + j0 + lcol] = c[r];
            }
        }
    }
}

// ---------------------------------------------------------------------------
extern "C" void kernel_launch(void* const* d_in, const int* in_sizes, int n_in,
                              void* d_out, int out_size, void* d_ws, size_t ws_size,
                              hipStream_t stream) {
    const float* signals = (const float*)d_in[0];   // [B,C,N,T] f32
    const float* Wc      = (const float*)d_in[1];   // [T,T]     f32
    const float* alpha   = (const float*)d_in[2];   // [N]       f32
    float*       outp    = (float*)d_out;           // [B,C,N,T] f32

    float* kw   = (float*)d_ws;                     // B*C*T = 131072 floats (512 KB)
    float* attw = kw + (size_t)BB * CC * TT;        // B*C*C =  65536 floats (256 KB)

    ca_reduce_k<<<dim3(BB * CC), dim3(TT), 0, stream>>>(signals, alpha, kw);
    ca_scores_softmax<<<dim3(BB), dim3(256), 0, stream>>>(kw, Wc, attw);
    ca_mix_wmma<<<dim3(128, BB), dim3(256), 0, stream>>>(signals, attw, outp);
}
